// Ptr_Net_28123445854492
// MI455X (gfx1250) — compile-verified, matrix operand order checked by reading, not run on hardware
//
#include <hip/hip_runtime.h>
#include <hip/hip_bf16.h>

typedef __attribute__((ext_vector_type(16))) _Float16 v16h;
typedef __attribute__((ext_vector_type(8)))  float    v8f;

#define B_N 512
#define S_N 256
#define V_N 10
#define E_N 128
#define H_N 128
#define D_N 256
#define G_N 512

__device__ __forceinline__ float sigf(float x) { return 1.0f / (1.0f + __expf(-x)); }

// Map (row m, k-within-32) -> (lane, half-slot) for the 16-bit WMMA A-fragment layout.
// lane<16 holds K={0..7,16..23}, lane>=16 holds K={8..15,24..31} (ISA 7.12.2).
__device__ __forceinline__ void a_slot(int m, int k32, int& lane, int& slot) {
    int grp = k32 >> 3, q = k32 & 7;
    lane = m + ((grp & 1) ? 16 : 0);
    slot = q + ((grp & 2) ? 8 : 0);
}

__device__ __forceinline__ v8f wmma_f16(v16h a, v16h b, v8f c) {
    return __builtin_amdgcn_wmma_f32_16x16x32_f16(false, a, false, b, (short)0, c, false, false);
}

// ---------------------------------------------------------------------------
// Pack W (N x K, row-major f32) used as B-operand of X @ W^T into f16 WMMA
// B-fragments: tile (nt,kt) linearized nt*(K/32)+kt, within tile lane*16+slot.
// ---------------------------------------------------------------------------
__global__ void pack_b_kernel(const float* __restrict__ W, _Float16* __restrict__ out,
                              int N, int K) {
    int wavesPerBlock = blockDim.x >> 5;
    int wave = blockIdx.x * wavesPerBlock + (threadIdx.x >> 5);
    int lane = threadIdx.x & 31;
    int nTiles = N >> 4, kTiles = K >> 5;
    if (wave >= nTiles * kTiles) return;
    int nt = wave / kTiles, kt = wave % kTiles;
    int n = nt * 16 + (lane & 15);
    int kbase = kt * 32 + ((lane >= 16) ? 8 : 0);
    _Float16* o = out + ((size_t)wave * 32 + lane) * 16;
#pragma unroll
    for (int s = 0; s < 16; ++s) {
        int k = kbase + (s < 8 ? s : s + 8);
        o[s] = (_Float16)W[(size_t)n * K + k];
    }
}

// u = Wih @ W_embed  (rank-1 embedding folded into the input-gate vector)
__global__ void u_kernel(const float* __restrict__ Wih, const float* __restrict__ Wemb,
                         float* __restrict__ u) {
    int n = blockIdx.x * blockDim.x + threadIdx.x;
    if (n >= G_N) return;
    float a = 0.f;
    for (int e = 0; e < E_N; ++e) a += Wih[(size_t)n * E_N + e] * Wemb[e];
    u[n] = a;
}

// ---------------------------------------------------------------------------
// Encoder: one block = 16 batch rows, one direction. Whh fragments in VGPRs.
// ---------------------------------------------------------------------------
__global__ __launch_bounds__(256, 1) void encoder_kernel(
    const float* __restrict__ s_inputs,
    const _Float16* __restrict__ whhP_f, const _Float16* __restrict__ whhP_b,
    const float* __restrict__ u_f, const float* __restrict__ u_b,
    const float* __restrict__ b_f, const float* __restrict__ b_b,
    float* __restrict__ Enc, float* __restrict__ hT, float* __restrict__ cT) {
    __shared__ v16h aFrag[4][32];
    __shared__ float gbuf[16][G_N];
    __shared__ float cbuf[16][H_N];
    __shared__ float hbuf[16][H_N];

    const int tid = threadIdx.x, wave = tid >> 5, lane = tid & 31;
    const int dir = blockIdx.y;
    const int b0 = blockIdx.x * 16;
    const _Float16* whhP = dir ? whhP_b : whhP_f;
    const float* u = dir ? u_b : u_f;
    const float* bb = dir ? b_b : b_f;

    v16h wreg[4][4];
#pragma unroll
    for (int j = 0; j < 4; ++j)
#pragma unroll
        for (int kt = 0; kt < 4; ++kt)
            wreg[j][kt] = *(const v16h*)(whhP + ((size_t)((wave * 4 + j) * 4 + kt) * 32 + lane) * 16);

    for (int i = tid; i < 4 * 32; i += 256) {
        v16h z;
#pragma unroll
        for (int q = 0; q < 16; ++q) z[q] = (_Float16)0;
        aFrag[i >> 5][i & 31] = z;
    }
    for (int i = tid; i < 16 * H_N; i += 256) { cbuf[i >> 7][i & 127] = 0.f; hbuf[i >> 7][i & 127] = 0.f; }
    __syncthreads();

    for (int t = 0; t < S_N; ++t) {
        const int tt = dir ? (S_N - 1 - t) : t;
        v8f acc[4];
#pragma unroll
        for (int j = 0; j < 4; ++j)
#pragma unroll
            for (int r = 0; r < 8; ++r) acc[j][r] = 0.f;
#pragma unroll
        for (int kt = 0; kt < 4; ++kt) {
            v16h a = aFrag[kt][lane];
#pragma unroll
            for (int j = 0; j < 4; ++j) acc[j] = wmma_f16(a, wreg[j][kt], acc[j]);
        }
        const int rowOff = (lane >> 4) * 8, col0 = lane & 15;
#pragma unroll
        for (int j = 0; j < 4; ++j) {
            int nc = (wave * 4 + j) * 16 + col0;
#pragma unroll
            for (int r = 0; r < 8; ++r) gbuf[rowOff + r][nc] = acc[j][r];
        }
        __syncthreads();
        for (int idx = tid; idx < 16 * H_N; idx += 256) {
            int m = idx >> 7, hh = idx & 127;
            int bi = b0 + m;
            float x = s_inputs[((size_t)bi * S_N + tt) * 3];
            float gi = gbuf[m][hh]       + x * u[hh]       + bb[hh];
            float gf = gbuf[m][hh + 128] + x * u[hh + 128] + bb[hh + 128];
            float gg = gbuf[m][hh + 256] + x * u[hh + 256] + bb[hh + 256];
            float go = gbuf[m][hh + 384] + x * u[hh + 384] + bb[hh + 384];
            float c = sigf(gf) * cbuf[m][hh] + sigf(gi) * tanhf(gg);
            float h = sigf(go) * tanhf(c);
            cbuf[m][hh] = c; hbuf[m][hh] = h;
            int ln, sl; a_slot(m, hh & 31, ln, sl);
            ((_Float16*)&aFrag[hh >> 5][ln])[sl] = (_Float16)h;
            Enc[((size_t)bi * S_N + tt) * D_N + dir * H_N + hh] = h;
        }
        __syncthreads();
    }
    for (int idx = tid; idx < 16 * H_N; idx += 256) {
        int m = idx >> 7, hh = idx & 127;
        int bi = b0 + m;
        hT[((size_t)dir * B_N + bi) * H_N + hh] = hbuf[m][hh];
        cT[((size_t)dir * B_N + bi) * H_N + hh] = cbuf[m][hh];
    }
}

// ---------------------------------------------------------------------------
// Fused dual projection: C1 = A@W_ref^T, C2 = A@W_ref2^T (A read once).
// One 16x16 output tile (for BOTH outputs) per wave.
// B fragments staged to LDS via async-to-LDS when available.
// grid.x = M/128, grid.y = 16 (nTile), block = 8 waves.
// ---------------------------------------------------------------------------
__global__ __launch_bounds__(256, 1) void gemm_proj2_kernel(
    const float* __restrict__ A,
    const _Float16* __restrict__ Bp1, const _Float16* __restrict__ Bp2,
    float* __restrict__ C1, float* __restrict__ C2) {
    __shared__ v16h bFrag[2][8][32];
    const int tid = threadIdx.x, wave = tid >> 5, lane = tid & 31;
    const int nTile = blockIdx.y;
    const int mTile = blockIdx.x * 8 + wave;
    {
        const _Float16* s1 = Bp1 + (size_t)nTile * 4096;  // 8 k-tiles x 512 halves, contiguous
        const _Float16* s2 = Bp2 + (size_t)nTile * 4096;
        _Float16* l = (_Float16*)bFrag;
#if __has_builtin(__builtin_amdgcn_global_load_async_to_lds_b128)
        // Signature (per compiler diagnostics):
        //   (v4i __device__(AS1)* src, v4i __shared__(AS3)* dst, imm offset, imm cpol)
        typedef int v4i_b128 __attribute__((vector_size(16)));
        typedef __attribute__((address_space(1))) v4i_b128* gas1_b128;
        typedef __attribute__((address_space(3))) v4i_b128* las3_b128;
        for (int i = tid; i < 512; i += 256) {
            __builtin_amdgcn_global_load_async_to_lds_b128(
                (gas1_b128)(uintptr_t)(s1 + (size_t)i * 8),
                (las3_b128)(uintptr_t)(l + (size_t)i * 8), 0, 0);
            __builtin_amdgcn_global_load_async_to_lds_b128(
                (gas1_b128)(uintptr_t)(s2 + (size_t)i * 8),
                (las3_b128)(uintptr_t)(l + 4096 + (size_t)i * 8), 0, 0);
        }
#if __has_builtin(__builtin_amdgcn_s_wait_asynccnt)
        __builtin_amdgcn_s_wait_asynccnt(0);
#else
        asm volatile("s_wait_asynccnt 0x0" ::: "memory");
#endif
#else
        v16h* lv = (v16h*)bFrag;
        for (int i = tid; i < 512; i += 256) {
            int half = i >> 8, idx = i & 255;
            const _Float16* s = half ? s2 : s1;
            lv[half * 256 + idx] = *(const v16h*)(s + (size_t)idx * 16);
        }
#endif
    }
    __syncthreads();
    const int mrow = lane & 15;
    const int khalf = (lane >> 4) ? 8 : 0;
    v8f acc1, acc2;
#pragma unroll
    for (int r = 0; r < 8; ++r) { acc1[r] = 0.f; acc2[r] = 0.f; }
    const float* Arow = A + ((size_t)(mTile * 16 + mrow)) * D_N;
#pragma unroll
    for (int kt = 0; kt < 8; ++kt) {
        v16h a;
        int kb = kt * 32 + khalf;
#pragma unroll
        for (int s = 0; s < 16; ++s) {
            int k = kb + (s < 8 ? s : s + 8);
            a[s] = (_Float16)Arow[k];
        }
        acc1 = wmma_f16(a, bFrag[0][kt][lane], acc1);
        acc2 = wmma_f16(a, bFrag[1][kt][lane], acc2);
    }
    const int rowOff = (lane >> 4) * 8;
#pragma unroll
    for (int r = 0; r < 8; ++r) {
        size_t o = ((size_t)(mTile * 16 + rowOff + r)) * D_N + nTile * 16 + (lane & 15);
        C1[o] = acc1[r];
        C2[o] = acc2[r];
    }
}

// ---------------------------------------------------------------------------
// Decoder: one block = 16 batch rows, loops all V=10 steps locally.
// ---------------------------------------------------------------------------
__global__ __launch_bounds__(256, 1) void decoder_kernel(
    const int* __restrict__ s_node_indexes,
    const float* __restrict__ s_inputs,
    const float* __restrict__ v_input,
    const _Float16* __restrict__ wihP_f, const _Float16* __restrict__ wihP_b,
    const _Float16* __restrict__ whhP_f, const _Float16* __restrict__ whhP_b,
    const float* __restrict__ bdf, const float* __restrict__ bdb,
    const _Float16* __restrict__ wqP, const _Float16* __restrict__ wq2P,
    const float* __restrict__ vw, const float* __restrict__ v2w,
    const float* __restrict__ refE, const float* __restrict__ ref2E,
    const float* __restrict__ Enc,
    const float* __restrict__ hT, const float* __restrict__ cT,
    float* __restrict__ out) {
    __shared__ v16h aIn[8][32];     // dec_in fragment (K=256)
    __shared__ v16h aOut[8][32];    // dec_out fragment (K=256)
    __shared__ v16h aHf[4][32];     // h fwd fragment (K=128)
    __shared__ v16h aHb[4][32];     // h bwd fragment
    __shared__ float gq[16][G_N];   // gates, then q | q2
    __shared__ float sbuf[16][S_N]; // ow, then att
    __shared__ float cf[16][H_N], cb[16][H_N];
    __shared__ float vwS[D_N], v2wS[D_N];
    __shared__ int selHist[V_N][16];
    __shared__ unsigned playedBits[16][8];

    const int tid = threadIdx.x, wave = tid >> 5, lane = tid & 31;
    const int b0 = blockIdx.x * 16;

    for (int i = tid; i < D_N; i += 256) { vwS[i] = vw[i]; v2wS[i] = v2w[i]; }
    for (int i = tid; i < 16 * 8; i += 256) playedBits[i >> 3][i & 7] = 0u;
    {
        v16h z;
#pragma unroll
        for (int q = 0; q < 16; ++q) z[q] = (_Float16)0;
        for (int i = tid; i < 8 * 32; i += 256) aIn[i >> 5][i & 31] = z;
    }
    for (int idx = tid; idx < 16 * H_N; idx += 256) {
        int m = idx >> 7, hh = idx & 127, bi = b0 + m;
        float hf0 = hT[((size_t)0 * B_N + bi) * H_N + hh];
        float hb0 = hT[((size_t)1 * B_N + bi) * H_N + hh];
        cf[m][hh] = cT[((size_t)0 * B_N + bi) * H_N + hh];
        cb[m][hh] = cT[((size_t)1 * B_N + bi) * H_N + hh];
        int ln, sl; a_slot(m, hh & 31, ln, sl);
        ((_Float16*)&aHf[hh >> 5][ln])[sl] = (_Float16)hf0;
        ((_Float16*)&aHb[hh >> 5][ln])[sl] = (_Float16)hb0;
    }
    __syncthreads();

    float* owOut = out + 2 * (size_t)B_N * V_N;

    for (int v = 0; v < V_N; ++v) {
        float vcpu = v_input[v * 3];
        // ---- two LSTM cells (fwd then bwd) ----
        for (int pass = 0; pass < 2; ++pass) {
            const _Float16* wih = pass ? wihP_b : wihP_f;
            const _Float16* whh = pass ? whhP_b : whhP_f;
            const float* bd = pass ? bdb : bdf;
            v8f acc[4];
#pragma unroll
            for (int j = 0; j < 4; ++j)
#pragma unroll
                for (int r = 0; r < 8; ++r) acc[j][r] = 0.f;
#pragma unroll
            for (int kt = 0; kt < 8; ++kt) {
                v16h a = aIn[kt][lane];
#pragma unroll
                for (int j = 0; j < 4; ++j) {
                    int nt = wave * 4 + j;
                    v16h bv = *(const v16h*)(wih + ((size_t)(nt * 8 + kt) * 32 + lane) * 16);
                    acc[j] = wmma_f16(a, bv, acc[j]);
                }
            }
#pragma unroll
            for (int kt = 0; kt < 4; ++kt) {
                v16h a = pass ? aHb[kt][lane] : aHf[kt][lane];
#pragma unroll
                for (int j = 0; j < 4; ++j) {
                    int nt = wave * 4 + j;
                    v16h bv = *(const v16h*)(whh + ((size_t)(nt * 4 + kt) * 32 + lane) * 16);
                    acc[j] = wmma_f16(a, bv, acc[j]);
                }
            }
            int rowOff = (lane >> 4) * 8, col0 = lane & 15;
#pragma unroll
            for (int j = 0; j < 4; ++j) {
                int nc = (wave * 4 + j) * 16 + col0;
#pragma unroll
                for (int r = 0; r < 8; ++r) gq[rowOff + r][nc] = acc[j][r];
            }
            __syncthreads();
            for (int idx = tid; idx < 16 * H_N; idx += 256) {
                int m = idx >> 7, hh = idx & 127;
                float gi = gq[m][hh] + bd[hh];
                float gf = gq[m][hh + 128] + bd[hh + 128];
                float gg = gq[m][hh + 256] + bd[hh + 256];
                float go = gq[m][hh + 384] + bd[hh + 384];
                float cOld = pass ? cb[m][hh] : cf[m][hh];
                float c = sigf(gf) * cOld + sigf(gi) * tanhf(gg);
                float h = sigf(go) * tanhf(c);
                if (pass) cb[m][hh] = c; else cf[m][hh] = c;
                int ln, sl; a_slot(m, hh & 31, ln, sl);
                if (pass) ((_Float16*)&aHb[hh >> 5][ln])[sl] = (_Float16)h;
                else      ((_Float16*)&aHf[hh >> 5][ln])[sl] = (_Float16)h;
                int d = pass ? hh + 128 : hh;
                int ln2, sl2; a_slot(m, d & 31, ln2, sl2);
                ((_Float16*)&aOut[d >> 5][ln2])[sl2] = (_Float16)h;
            }
            __syncthreads();
        }
        // ---- q = dec_out@Wq^T (gq cols 0..255), q2 = dec_in@Wq2^T (cols 256..511) ----
        {
            v8f acc[4];
#pragma unroll
            for (int j = 0; j < 4; ++j)
#pragma unroll
                for (int r = 0; r < 8; ++r) acc[j][r] = 0.f;
#pragma unroll
            for (int kt = 0; kt < 8; ++kt) {
                v16h ao = aOut[kt][lane];
                v16h ai = aIn[kt][lane];
#pragma unroll
                for (int j = 0; j < 2; ++j) {
                    int nt = wave * 2 + j;
                    v16h bq  = *(const v16h*)(wqP  + ((size_t)(nt * 8 + kt) * 32 + lane) * 16);
                    v16h bq2 = *(const v16h*)(wq2P + ((size_t)(nt * 8 + kt) * 32 + lane) * 16);
                    acc[j]     = wmma_f16(ao, bq,  acc[j]);
                    acc[2 + j] = wmma_f16(ai, bq2, acc[2 + j]);
                }
            }
            int rowOff = (lane >> 4) * 8, col0 = lane & 15;
#pragma unroll
            for (int j = 0; j < 2; ++j) {
                int nc = (wave * 2 + j) * 16 + col0;
#pragma unroll
                for (int r = 0; r < 8; ++r) {
                    gq[rowOff + r][nc] = acc[j][r];
                    gq[rowOff + r][256 + nc] = acc[2 + j][r];
                }
            }
            __syncthreads();
        }
        // ---- ow logits + penalty ----
        for (int p = tid; p < 16 * S_N; p += 256) {
            int m = p >> 8, s = p & 255, bi = b0 + m;
            const float* rr = refE + ((size_t)bi * S_N + s) * D_N;
            if (m < 15)
                __builtin_prefetch(refE + ((size_t)(bi + 1) * S_N + s) * D_N, 0, 1);
            float acc = 0.f;
            for (int d = 0; d < D_N; ++d) acc += tanhf(rr[d] + gq[m][d]) * vwS[d];
            float cannot = ((s_inputs[((size_t)bi * S_N + s) * 3] < vcpu) ? 1.f : 0.f)
                         + (((playedBits[m][s >> 5] >> (s & 31)) & 1u) ? 1.f : 0.f);
            float o = acc - 1.0e6f * cannot;
            sbuf[m][s] = o;
            owOut[((size_t)v * B_N + bi) * S_N + s] = o;
        }
        __syncthreads();
        // ---- argmax + played update ----
        if (tid < 16) {
            int m = tid; float best = sbuf[m][0]; int bs = 0;
            for (int s = 1; s < S_N; ++s) { float x = sbuf[m][s]; if (x > best) { best = x; bs = s; } }
            selHist[v][m] = bs;
            playedBits[m][bs >> 5] |= (1u << (bs & 31));
        }
        __syncthreads();
        // ---- attention logits ----
        for (int p = tid; p < 16 * S_N; p += 256) {
            int m = p >> 8, s = p & 255, bi = b0 + m;
            const float* rr = ref2E + ((size_t)bi * S_N + s) * D_N;
            if (m < 15)
                __builtin_prefetch(ref2E + ((size_t)(bi + 1) * S_N + s) * D_N, 0, 1);
            float acc = 0.f;
            for (int d = 0; d < D_N; ++d) acc += tanhf(rr[d] + gq[m][256 + d]) * v2wS[d];
            sbuf[m][s] = acc;
        }
        __syncthreads();
        if (tid < 16) {
            int m = tid; float mx = sbuf[m][0];
            for (int s = 1; s < S_N; ++s) mx = fmaxf(mx, sbuf[m][s]);
            float sum = 0.f;
            for (int s = 0; s < S_N; ++s) { float e = __expf(sbuf[m][s] - mx); sbuf[m][s] = e; sum += e; }
            float inv = 1.f / sum;
            for (int s = 0; s < S_N; ++s) sbuf[m][s] *= inv;
        }
        __syncthreads();
        // ---- new_in = att @ Enc, packed straight into the aIn fragment ----
        {
            int d = tid;  // D_N == 256 == blockDim
            for (int m = 0; m < 16; ++m) {
                int bi = b0 + m;
                float acc = 0.f;
                const float* ep = Enc + ((size_t)bi * S_N) * D_N + d;
                for (int s = 0; s < S_N; ++s) acc += sbuf[m][s] * ep[(size_t)s * D_N];
                int ln, sl; a_slot(m, d & 31, ln, sl);
                ((_Float16*)&aIn[d >> 5][ln])[sl] = (_Float16)acc;
            }
        }
        __syncthreads();
    }
    // ---- final index outputs (stored as float values) ----
    for (int p = tid; p < 16 * V_N; p += 256) {
        int m = p & 15, v = p >> 4, bi = b0 + m;
        int sv = selHist[v][m];
        out[(size_t)bi * V_N + v] = (float)s_node_indexes[(size_t)bi * S_N + sv];
        out[(size_t)B_N * V_N + (size_t)bi * V_N + v] = (float)sv;
    }
}

// ---------------------------------------------------------------------------
extern "C" void kernel_launch(void* const* d_in, const int* in_sizes, int n_in,
                              void* d_out, int out_size, void* d_ws, size_t ws_size,
                              hipStream_t stream) {
    (void)in_sizes; (void)n_in; (void)out_size; (void)ws_size;
    const int*   s_node_indexes = (const int*)d_in[0];
    const float* s_inputs  = (const float*)d_in[1];
    const float* v_input   = (const float*)d_in[2];
    const float* W_embed   = (const float*)d_in[3];
    const float* enc_Wih_f = (const float*)d_in[4];
    const float* enc_Whh_f = (const float*)d_in[5];
    const float* enc_b_f   = (const float*)d_in[6];
    const float* enc_Wih_b = (const float*)d_in[7];
    const float* enc_Whh_b = (const float*)d_in[8];
    const float* enc_b_b   = (const float*)d_in[9];
    const float* dec_Wih_f = (const float*)d_in[10];
    const float* dec_Whh_f = (const float*)d_in[11];
    const float* dec_b_f   = (const float*)d_in[12];
    const float* dec_Wih_b = (const float*)d_in[13];
    const float* dec_Whh_b = (const float*)d_in[14];
    const float* dec_b_b   = (const float*)d_in[15];
    const float* W_ref     = (const float*)d_in[16];
    const float* W_q       = (const float*)d_in[17];
    const float* v_w       = (const float*)d_in[18];
    const float* W_ref2    = (const float*)d_in[19];
    const float* W_q2      = (const float*)d_in[20];
    const float* v2_w      = (const float*)d_in[21];
    float* out = (float*)d_out;

    char* w = (char*)d_ws;
    size_t off = 0;
    auto alloc = [&](size_t bytes) -> char* {
        char* p = w + off;
        off += (bytes + 255) & ~(size_t)255;
        return p;
    };
    _Float16* pEncWhhF = (_Float16*)alloc((size_t)G_N * H_N * 2);
    _Float16* pEncWhhB = (_Float16*)alloc((size_t)G_N * H_N * 2);
    _Float16* pDecWihF = (_Float16*)alloc((size_t)G_N * D_N * 2);
    _Float16* pDecWihB = (_Float16*)alloc((size_t)G_N * D_N * 2);
    _Float16* pDecWhhF = (_Float16*)alloc((size_t)G_N * H_N * 2);
    _Float16* pDecWhhB = (_Float16*)alloc((size_t)G_N * H_N * 2);
    _Float16* pWref  = (_Float16*)alloc((size_t)D_N * D_N * 2);
    _Float16* pWref2 = (_Float16*)alloc((size_t)D_N * D_N * 2);
    _Float16* pWq    = (_Float16*)alloc((size_t)D_N * D_N * 2);
    _Float16* pWq2   = (_Float16*)alloc((size_t)D_N * D_N * 2);
    float* uF  = (float*)alloc((size_t)G_N * 4);
    float* uB  = (float*)alloc((size_t)G_N * 4);
    float* Enc = (float*)alloc((size_t)B_N * S_N * D_N * 4);
    float* rE  = (float*)alloc((size_t)B_N * S_N * D_N * 4);
    float* r2E = (float*)alloc((size_t)B_N * S_N * D_N * 4);
    float* hT  = (float*)alloc((size_t)2 * B_N * H_N * 4);
    float* cT  = (float*)alloc((size_t)2 * B_N * H_N * 4);

    auto pack = [&](const float* W, _Float16* o, int N, int K) {
        int tiles = (N / 16) * (K / 32);
        pack_b_kernel<<<(tiles + 7) / 8, 256, 0, stream>>>(W, o, N, K);
    };
    pack(enc_Whh_f, pEncWhhF, G_N, H_N);
    pack(enc_Whh_b, pEncWhhB, G_N, H_N);
    pack(dec_Wih_f, pDecWihF, G_N, D_N);
    pack(dec_Wih_b, pDecWihB, G_N, D_N);
    pack(dec_Whh_f, pDecWhhF, G_N, H_N);
    pack(dec_Whh_b, pDecWhhB, G_N, H_N);
    pack(W_ref,  pWref,  D_N, D_N);
    pack(W_ref2, pWref2, D_N, D_N);
    pack(W_q,    pWq,    D_N, D_N);
    pack(W_q2,   pWq2,   D_N, D_N);

    u_kernel<<<2, 256, 0, stream>>>(enc_Wih_f, W_embed, uF);
    u_kernel<<<2, 256, 0, stream>>>(enc_Wih_b, W_embed, uB);

    encoder_kernel<<<dim3(B_N / 16, 2), 256, 0, stream>>>(
        s_inputs, pEncWhhF, pEncWhhB, uF, uB, enc_b_f, enc_b_b, Enc, hT, cT);

    // M = B*S = 131072 rows -> grid.x = M/128 = 1024, grid.y = N/16 = 16
    gemm_proj2_kernel<<<dim3((B_N * S_N) / 128, D_N / 16), 256, 0, stream>>>(
        Enc, pWref, pWref2, rE, r2E);

    decoder_kernel<<<dim3(B_N / 16), 256, 0, stream>>>(
        s_node_indexes, s_inputs, v_input,
        pDecWihF, pDecWihB, pDecWhhF, pDecWhhB, dec_b_f, dec_b_b,
        pWq, pWq2, v_w, v2_w, rE, r2E, Enc, hT, cT, out);
}
